// MultiheadAttention_61100204753668
// MI455X (gfx1250) — matrix operand order repaired and checked by reading, not compiled, not run on hardware
//
#include <hip/hip_runtime.h>
#include <hip/hip_bf16.h>

// ---------------------------------------------------------------------------
// MHA forward for MI455X (gfx1250).
// All matmuls through v_wmma_f32_16x16x32_bf16; LDS staging via
// GLOBAL_LOAD_ASYNC_TO_LDS_B128 (ASYNCcnt) when the builtin is available.
// ---------------------------------------------------------------------------

#define BATCH  2
#define SEQ    2048
#define DMODEL 1024
#define NHEAD  16
#define HDIM   64

typedef __attribute__((ext_vector_type(16))) __bf16 v16bf;
typedef __attribute__((ext_vector_type(8)))  float  v8f;
typedef __attribute__((ext_vector_type(4)))  int    v4i_t;

#if defined(__has_builtin)
#  if __has_builtin(__builtin_amdgcn_global_load_async_to_lds_b128)
#    define HAVE_ASYNC_LDS 1
#  endif
#endif
#ifndef HAVE_ASYNC_LDS
#  define HAVE_ASYNC_LDS 0
#endif

__device__ __forceinline__ unsigned short f2bf_u(float f) {
    union { float f; unsigned int u; } c; c.f = f;
    unsigned int u = c.u;
    u += 0x7FFFu + ((u >> 16) & 1u);          // round-to-nearest-even
    return (unsigned short)(u >> 16);
}
__device__ __forceinline__ __bf16 u2bf(unsigned short s) {
    union { unsigned short u; __bf16 b; } c; c.u = s; return c.b;
}

// 16-byte global -> LDS copy: async DMA path on CDNA5, uint4 fallback.
// Builtin prototype (from hipcc diagnostic): (AS1 v4i*, AS3 v4i*, Ii, Ii).
__device__ __forceinline__ void g2lds_16B(unsigned short* lds_dst,
                                          const unsigned short* gsrc) {
#if HAVE_ASYNC_LDS
    __builtin_amdgcn_global_load_async_to_lds_b128(
        (__attribute__((address_space(1))) v4i_t*)gsrc,
        (__attribute__((address_space(3))) v4i_t*)lds_dst, 0, 0);
#else
    *(uint4*)lds_dst = *(const uint4*)gsrc;
#endif
}
// Wait for this wave's async stores to LDS, then workgroup barrier.
__device__ __forceinline__ void wait_lds_stage() {
#if HAVE_ASYNC_LDS
    asm volatile("s_wait_asynccnt 0" ::: "memory");
#endif
    __syncthreads();
}

// ---- WMMA fragment loaders (contiguous k order, same for A and B, wave32) --------------------
// A fragment: 16x32 (MxK) bf16. lane<16 holds K {0..15}, lane>=16 holds
// K {16..31}; M = lane & 15.
__device__ __forceinline__ v16bf load_A_frag(const unsigned short* lds,
                                             int m0, int k0, int ld) {
    const int lane  = threadIdx.x & 31;
    const int m     = m0 + (lane & 15);
    const int kbase = k0 + ((lane >> 4) << 4);
    v16bf a;
#pragma unroll
    for (int h = 0; h < 16; ++h) {
        int k = kbase + h;
        a[h] = u2bf(lds[m * ld + k]);
    }
    return a;
}
// B fragment where element(k,n) is stored at lds[n*ld + k] (n-major rows):
// used for W (N,K), K-block (n,d) and transposed-V (d,key).
__device__ __forceinline__ v16bf load_B_nk(const unsigned short* lds,
                                           int n0, int k0, int ld) {
    const int lane = threadIdx.x & 31;
    const int n    = n0 + (lane & 15);
    const int kb   = k0 + ((lane >> 4) << 4);
    v16bf b;
#pragma unroll
    for (int h = 0; h < 16; ++h) b[h] = u2bf(lds[n * ld + kb + h]);
    return b;
}

// ---------------------------------------------------------------------------
// Elementwise f32 -> bf16 cast (one pass; removes conversions from GEMM loop)
// ---------------------------------------------------------------------------
__global__ __launch_bounds__(256)
void cast_f32_bf16(const float* __restrict__ src, unsigned short* __restrict__ dst,
                   int n4) {   // n4 = n/4, n divisible by 4
    int i = blockIdx.x * 256 + threadIdx.x;
    if (i < n4) {
        float4 f = ((const float4*)src)[i];
        uint2 o;
        o.x = (unsigned)f2bf_u(f.x) | ((unsigned)f2bf_u(f.y) << 16);
        o.y = (unsigned)f2bf_u(f.z) | ((unsigned)f2bf_u(f.w) << 16);
        ((uint2*)dst)[i] = o;
    }
}

// ---------------------------------------------------------------------------
// GEMM: out[m,n] = sum_k A[m,k] * W[n,k] + bias[n]
// A: bf16 (M,K) row-major; W: bf16 (N,K) row-major; output bf16 or f32.
// Block tile 128x128, 8 waves (2x4), wave tile 64x32, K stepped by 64.
// ---------------------------------------------------------------------------
#define BM  128
#define BN  128
#define BKK 64
#define LDT 72   // LDS pitch in shorts (64 + 8 pad; 144 B rows keep 16B align)

__global__ __launch_bounds__(256)
void gemm_xwT_bias(const unsigned short* __restrict__ A,
                   const unsigned short* __restrict__ W,
                   const float* __restrict__ bias,
                   unsigned short* __restrict__ dstb, float* __restrict__ dstf,
                   int M, int N, int K, int store_bf16) {
    __shared__ unsigned short As[BM * LDT];
    __shared__ unsigned short Ws[BN * LDT];

    const int tid  = threadIdx.x;
    const int lane = tid & 31;
    const int wave = tid >> 5;
    const int m0   = blockIdx.y * BM;
    const int n0   = blockIdx.x * BN;
    const int wm   = (wave >> 2) * 64;   // 0 / 64
    const int wn   = (wave & 3) * 32;    // 0 / 32 / 64 / 96

    v8f acc[4][2];
#pragma unroll
    for (int i = 0; i < 4; ++i)
#pragma unroll
        for (int j = 0; j < 2; ++j)
#pragma unroll
            for (int r = 0; r < 8; ++r) acc[i][j][r] = 0.0f;

    for (int k0 = 0; k0 < K; k0 += BKK) {
        // stage bf16 tiles: pure 16B copies (async DMA to LDS on CDNA5)
        for (int v = tid; v < (BM * BKK) / 8; v += 256) {   // 1024 vectors each
            int r = v >> 3, c8 = (v & 7) * 8;
            g2lds_16B(&As[r * LDT + c8], &A[(size_t)(m0 + r) * K + k0 + c8]);
            g2lds_16B(&Ws[r * LDT + c8], &W[(size_t)(n0 + r) * K + k0 + c8]);
        }
        if (k0 + BKK < K) {   // global_prefetch_b8 on the next K-tile
            __builtin_prefetch(&A[(size_t)(m0 + (tid >> 1)) * K + k0 + BKK + (tid & 1) * 32], 0, 1);
            __builtin_prefetch(&W[(size_t)(n0 + (tid >> 1)) * K + k0 + BKK + (tid & 1) * 32], 0, 1);
        }
        wait_lds_stage();

#pragma unroll
        for (int kk = 0; kk < BKK; kk += 32) {
            v16bf af[4], bf[2];
#pragma unroll
            for (int i = 0; i < 4; ++i) af[i] = load_A_frag(As, wm + 16 * i, kk, LDT);
#pragma unroll
            for (int j = 0; j < 2; ++j) bf[j] = load_B_nk(Ws, wn + 16 * j, kk, LDT);
#pragma unroll
            for (int i = 0; i < 4; ++i)
#pragma unroll
                for (int j = 0; j < 2; ++j)
                    acc[i][j] = __builtin_amdgcn_wmma_f32_16x16x32_bf16(
                        false, af[i], false, bf[j], (short)0, acc[i][j], false, false);
        }
        __syncthreads();
    }

    // epilogue: bias add, store (C/D layout: M = r + 8*(lane>>4), N = lane&15)
#pragma unroll
    for (int i = 0; i < 4; ++i)
#pragma unroll
        for (int j = 0; j < 2; ++j) {
            int rr = m0 + wm + 16 * i + ((lane >> 4) << 3);
            int cc = n0 + wn + 16 * j + (lane & 15);
            float bv = bias[cc];
#pragma unroll
            for (int r = 0; r < 8; ++r) {
                float v = acc[i][j][r] + bv;
                if (store_bf16) dstb[(size_t)(rr + r) * N + cc] = f2bf_u(v);
                else            dstf[(size_t)(rr + r) * N + cc] = v;
            }
        }
}

// ---------------------------------------------------------------------------
// Flash attention: one block per (batch, head, 64-row q-tile). Online softmax
// over 64-key blocks; QK^T and PV via bf16 WMMA. V staged transposed so the
// PV B-operand reads contiguous LDS (b128). Key mask additive -1e9.
// ---------------------------------------------------------------------------
__global__ __launch_bounds__(256)
void flash_attn(const unsigned short* __restrict__ Qb,
                const unsigned short* __restrict__ Kb,
                const unsigned short* __restrict__ Vb,
                const int* __restrict__ mask,
                unsigned short* __restrict__ Xout) {   // (B,S,D) bf16
    __shared__ unsigned short Qs[64 * LDT];
    __shared__ unsigned short Ks[64 * LDT];
    __shared__ unsigned short Vt[64 * LDT];   // transposed: [d][key]
    __shared__ unsigned short Ps[64 * LDT];
    __shared__ float Ss[64 * 68];
    __shared__ float mrow[64], lrow[64], srow[64], kmsk[64];

    const int tid  = threadIdx.x;
    const int lane = tid & 31;
    const int wave = tid >> 5;
    const int bid  = blockIdx.x;
    const int qt   = bid & (SEQ / 64 - 1);          // 32 q-tiles
    const int h    = (bid >> 5) & (NHEAD - 1);
    const int b    = bid >> 9;
    const int q0   = qt * 64;
    const int hc   = h * HDIM;

    // Q tile (64 rows x 64 head-dims), async 16B copies
    for (int v = tid; v < 64 * 8; v += 256) {
        int r = v >> 3, c8 = (v & 7) * 8;
        g2lds_16B(&Qs[r * LDT + c8],
                  &Qb[(size_t)(b * SEQ + q0 + r) * DMODEL + hc + c8]);
    }
    if (tid < 64) { mrow[tid] = -3.0e38f; lrow[tid] = 0.0f; }

    // wave -> fragment mapping: 8 waves own the 4x4 grid of 16x16 frags
    const int mi  = (wave >> 1) * 16;   // row base in tile
    const int njb = (wave & 1) * 32;    // col base in tile
    v8f oacc[2];
#pragma unroll
    for (int j = 0; j < 2; ++j)
#pragma unroll
        for (int r = 0; r < 8; ++r) oacc[j][r] = 0.0f;

    for (int kb = 0; kb < SEQ; kb += 64) {
        __syncthreads();    // previous iteration done with Ks/Vt/Ps/Ss
        // K rows: async 16B copies
        for (int v = tid; v < 64 * 8; v += 256) {
            int r = v >> 3, c8 = (v & 7) * 8;
            g2lds_16B(&Ks[r * LDT + c8],
                      &Kb[(size_t)(b * SEQ + kb + r) * DMODEL + hc + c8]);
        }
        // V rows: read coalesced, store transposed (d-major)
        for (int v = tid; v < 64 * 16; v += 256) {
            int r = v >> 4, c4 = (v & 15) * 4;
            uint2 d = *(const uint2*)&Vb[(size_t)(b * SEQ + kb + r) * DMODEL + hc + c4];
            Vt[(c4 + 0) * LDT + r] = (unsigned short)(d.x & 0xffffu);
            Vt[(c4 + 1) * LDT + r] = (unsigned short)(d.x >> 16);
            Vt[(c4 + 2) * LDT + r] = (unsigned short)(d.y & 0xffffu);
            Vt[(c4 + 3) * LDT + r] = (unsigned short)(d.y >> 16);
        }
        if (tid < 64) kmsk[tid] = (mask[b * SEQ + kb + tid] == 0) ? -1.0e9f : 0.0f;
        wait_lds_stage();

        // S = Q K^T / sqrt(HD) + mask  (K^T realized by n-major LDS indexing)
#pragma unroll
        for (int j = 0; j < 2; ++j) {
            v8f s;
#pragma unroll
            for (int r = 0; r < 8; ++r) s[r] = 0.0f;
#pragma unroll
            for (int kk = 0; kk < HDIM; kk += 32) {
                v16bf a  = load_A_frag(Qs, mi, kk, LDT);
                v16bf bb = load_B_nk(Ks, njb + 16 * j, kk, LDT);
                s = __builtin_amdgcn_wmma_f32_16x16x32_bf16(
                        false, a, false, bb, (short)0, s, false, false);
            }
            int cc = njb + 16 * j + (lane & 15);
            int rr = mi + ((lane >> 4) << 3);
            float km = kmsk[cc];
#pragma unroll
            for (int r = 0; r < 8; ++r) Ss[(rr + r) * 68 + cc] = s[r] * 0.125f + km;
        }
        __syncthreads();

        // online softmax: lane t owns row t
        if (tid < 64) {
            const float* sp = &Ss[tid * 68];
            float mold = mrow[tid];
            float mx = mold;
#pragma unroll 8
            for (int c = 0; c < 64; ++c) mx = fmaxf(mx, sp[c]);
            float corr = __expf(mold - mx);
            float sum = 0.0f;
#pragma unroll 8
            for (int c = 0; c < 64; ++c) {
                float p = __expf(sp[c] - mx);
                sum += p;
                Ps[tid * LDT + c] = f2bf_u(p);
            }
            mrow[tid] = mx;
            lrow[tid] = lrow[tid] * corr + sum;
            srow[tid] = corr;
        }
        __syncthreads();

        // O = O*corr + P V   (Vt is d-major: element(k=key, N=d) = Vt[d*ld+key])
        {
            int rr = mi + ((lane >> 4) << 3);
#pragma unroll
            for (int j = 0; j < 2; ++j) {
#pragma unroll
                for (int r = 0; r < 8; ++r) oacc[j][r] *= srow[rr + r];
#pragma unroll
                for (int kk = 0; kk < 64; kk += 32) {
                    v16bf a  = load_A_frag(Ps, mi, kk, LDT);
                    v16bf bb = load_B_nk(Vt, njb + 16 * j, kk, LDT);
                    oacc[j] = __builtin_amdgcn_wmma_f32_16x16x32_bf16(
                                  false, a, false, bb, (short)0, oacc[j], false, false);
                }
            }
        }
    }
    __syncthreads();

    // finalize: divide by row sums, write bf16 (B,S,D) for output projection
    {
        int rr = mi + ((lane >> 4) << 3);
#pragma unroll
        for (int j = 0; j < 2; ++j) {
            int cc = njb + 16 * j + (lane & 15);
#pragma unroll
            for (int r = 0; r < 8; ++r) {
                float inv = 1.0f / lrow[rr + r];
                Xout[(size_t)(b * SEQ + q0 + rr + r) * DMODEL + hc + cc] =
                    f2bf_u(oacc[j][r] * inv);
            }
        }
    }
}

// ---------------------------------------------------------------------------
extern "C" void kernel_launch(void* const* d_in, const int* in_sizes, int n_in,
                              void* d_out, int out_size, void* d_ws, size_t ws_size,
                              hipStream_t stream) {
    const float* query = (const float*)d_in[0];
    const float* key   = (const float*)d_in[1];
    const float* value = (const float*)d_in[2];
    const int*   mask  = (const int*)d_in[3];
    const float* Wq = (const float*)d_in[4];
    const float* bq = (const float*)d_in[5];
    const float* Wk = (const float*)d_in[6];
    const float* bk = (const float*)d_in[7];
    const float* Wv = (const float*)d_in[8];
    const float* bv = (const float*)d_in[9];
    const float* Wo = (const float*)d_in[10];
    const float* bo = (const float*)d_in[11];
    float* out = (float*)d_out;

    const int M    = BATCH * SEQ;                    // 4096
    const int NACT = M * DMODEL;                     // 4,194,304
    const int NWT  = DMODEL * DMODEL;                // 1,048,576

    char* ws = (char*)d_ws;
    size_t off = 0;
    unsigned short* qin = (unsigned short*)(ws + off); off += (size_t)NACT * 2;
    unsigned short* kin = (unsigned short*)(ws + off); off += (size_t)NACT * 2;
    unsigned short* vin = (unsigned short*)(ws + off); off += (size_t)NACT * 2;
    unsigned short* wqb = (unsigned short*)(ws + off); off += (size_t)NWT * 2;
    unsigned short* wkb = (unsigned short*)(ws + off); off += (size_t)NWT * 2;
    unsigned short* wvb = (unsigned short*)(ws + off); off += (size_t)NWT * 2;
    unsigned short* wob = (unsigned short*)(ws + off); off += (size_t)NWT * 2;
    unsigned short* qbf = (unsigned short*)(ws + off); off += (size_t)NACT * 2;
    unsigned short* kbf = (unsigned short*)(ws + off); off += (size_t)NACT * 2;
    unsigned short* vbf = (unsigned short*)(ws + off); off += (size_t)NACT * 2;
    unsigned short* xbf = (unsigned short*)(ws + off); off += (size_t)NACT * 2;
    (void)ws_size; (void)in_sizes; (void)n_in; (void)out_size;

    dim3 blk(256);
    dim3 gact((NACT / 4 + 255) / 256);
    dim3 gwt((NWT / 4 + 255) / 256);

    cast_f32_bf16<<<gact, blk, 0, stream>>>(query, qin, NACT / 4);
    cast_f32_bf16<<<gact, blk, 0, stream>>>(key,   kin, NACT / 4);
    cast_f32_bf16<<<gact, blk, 0, stream>>>(value, vin, NACT / 4);
    cast_f32_bf16<<<gwt,  blk, 0, stream>>>(Wq, wqb, NWT / 4);
    cast_f32_bf16<<<gwt,  blk, 0, stream>>>(Wk, wkb, NWT / 4);
    cast_f32_bf16<<<gwt,  blk, 0, stream>>>(Wv, wvb, NWT / 4);
    cast_f32_bf16<<<gwt,  blk, 0, stream>>>(Wo, wob, NWT / 4);

    dim3 gg(DMODEL / BN, M / BM);                    // (8, 32)
    gemm_xwT_bias<<<gg, blk, 0, stream>>>(qin, wqb, bq, qbf, nullptr, M, DMODEL, DMODEL, 1);
    gemm_xwT_bias<<<gg, blk, 0, stream>>>(kin, wkb, bk, kbf, nullptr, M, DMODEL, DMODEL, 1);
    gemm_xwT_bias<<<gg, blk, 0, stream>>>(vin, wvb, bv, vbf, nullptr, M, DMODEL, DMODEL, 1);

    flash_attn<<<dim3(BATCH * NHEAD * (SEQ / 64)), blk, 0, stream>>>(qbf, kbf, vbf, mask, xbf);

    gemm_xwT_bias<<<gg, blk, 0, stream>>>(xbf, wob, bo, nullptr, out, M, DMODEL, DMODEL, 0);
}
